// AttentionHead_11519102288407
// MI455X (gfx1250) — matrix operand order repaired and checked by reading, not compiled, not run on hardware
//
#include <hip/hip_runtime.h>

// AttentionHead for MI455X (gfx1250): f16 WMMA projections + fused flash attention.
// Workspace layout (needs ~6.3 MB):
//   qh : _Float16 [B][T][H]        row-major, 2 MB
//   kh : _Float16 [B][T][H]        row-major, 2 MB
//   vT : _Float16 [B][H][T]        transposed so PV B-fragments are contiguous, 2 MB
//   Wt : _Float16 [3][H][C]        transposed f16 weights (q,k,v), 288 KB

#define B_ 8
#define T_ 2048
#define C_ 768
#define H_ 64

typedef __attribute__((ext_vector_type(16))) _Float16 v16h;
typedef __attribute__((ext_vector_type(8)))  _Float16 v8h;
typedef __attribute__((ext_vector_type(8)))  float    v8f;

#define WMMA_F16(a, b, c) \
  __builtin_amdgcn_wmma_f32_16x16x32_f16(false, (a), false, (b), (short)0, (c), false, false)

// A-matrix fragment (16x32 f16, MxK): lane L (m=L&15, half=L>>4)
// holds K = 16*half + {0..15} (contiguous, matching the B fragments).
// p points at row m's K-window base (16-byte aligned).
__device__ __forceinline__ v16h make_afrag(const _Float16* p, int hsel) {
  v8h c0 = *(const v8h*)(p + hsel * 16);
  v8h c1 = *(const v8h*)(p + hsel * 16 + 8);
  return __builtin_shufflevector(c0, c1, 0,1,2,3,4,5,6,7,8,9,10,11,12,13,14,15);
}
// B-matrix fragment (32x16 f16, KxN): lane L (n=L&15, half=L>>4) holds
// K = 16*half + {0..15} of column n => one contiguous 32-byte load when the
// source is column-major (contiguous in K).

// ---------------------------------------------------------------------------
// One-time weight conversion: Wt[which][h][c] = (f16) W[c][h].
// 3*C*H = 147456 elements; coalesced writes, L2-absorbed strided reads.
// ---------------------------------------------------------------------------
__global__ __launch_bounds__(256) void wcvt_kernel(
    const float* __restrict__ Wq, const float* __restrict__ Wk,
    const float* __restrict__ Wv, _Float16* __restrict__ Wt)
{
  const int idx = blockIdx.x * 256 + threadIdx.x;
  const int which = idx / (C_ * H_);
  const int rem   = idx % (C_ * H_);
  const int h = rem / C_, c = rem % C_;
  const float* __restrict__ W = (which == 0) ? Wq : (which == 1) ? Wk : Wv;
  Wt[idx] = (_Float16)W[(size_t)c * H_ + h];
}

// ---------------------------------------------------------------------------
// Projection: y = x @ W  (M=16384, K=768, N=64), f32 x -> f16 y.
// gridDim = (256, 3); blockIdx.y selects q/k/v; v is stored transposed.
// x is staged f32->f16 through double-buffered LDS (one barrier per K-chunk);
// W fragments come straight from global Wt (contiguous in K, L2-resident).
// ---------------------------------------------------------------------------
__global__ __launch_bounds__(256) void proj_kernel(
    const float* __restrict__ x, const _Float16* __restrict__ Wt,
    _Float16* __restrict__ qh, _Float16* __restrict__ kh, _Float16* __restrict__ vT)
{
  __shared__ _Float16 Axs[2][64 * 40];  // x tile ping-pong, padded stride 40
  __shared__ _Float16 Ots[64 * 72];     // V-tile transpose staging (which==2 only)

  const int which = blockIdx.y;
  const _Float16* __restrict__ Wb = Wt + (size_t)which * (C_ * H_);
  const int row0 = blockIdx.x * 64;               // token rows [row0, row0+64)
  const int tid  = threadIdx.x;
  const int wave = tid >> 5, lane = tid & 31;
  const int m = lane & 15, hsel = lane >> 4;
  const int wrow = (wave & 3) * 16;               // wave's output-row base
  const int wcol = (wave >> 2) * 32;              // wave's output-col base

  const int xr = tid >> 2, xc = (tid & 3) * 8;    // x-chunk loader: 64 rows x 32

  v8f acc0 = {}; v8f acc1 = {};

  // prologue: stage chunk 0
  {
    const float4 f0 = *(const float4*)(x + (size_t)(row0 + xr) * C_ + xc);
    const float4 f1 = *(const float4*)(x + (size_t)(row0 + xr) * C_ + xc + 4);
    v8h h;
    h[0] = (_Float16)f0.x; h[1] = (_Float16)f0.y; h[2] = (_Float16)f0.z; h[3] = (_Float16)f0.w;
    h[4] = (_Float16)f1.x; h[5] = (_Float16)f1.y; h[6] = (_Float16)f1.z; h[7] = (_Float16)f1.w;
    *(v8h*)&Axs[0][xr * 40 + xc] = h;
  }
  __syncthreads();

  int cur = 0;
  for (int k0 = 0; k0 < C_; k0 += 32) {
    // issue next chunk's global loads under this chunk's WMMAs
    v8h hnext;
    const bool more = (k0 + 32 < C_);
    if (more) {
      const float4 f0 = *(const float4*)(x + (size_t)(row0 + xr) * C_ + k0 + 32 + xc);
      const float4 f1 = *(const float4*)(x + (size_t)(row0 + xr) * C_ + k0 + 32 + xc + 4);
      hnext[0] = (_Float16)f0.x; hnext[1] = (_Float16)f0.y;
      hnext[2] = (_Float16)f0.z; hnext[3] = (_Float16)f0.w;
      hnext[4] = (_Float16)f1.x; hnext[5] = (_Float16)f1.y;
      hnext[6] = (_Float16)f1.z; hnext[7] = (_Float16)f1.w;
    }

    const v16h a  = make_afrag(&Axs[cur][(wrow + m) * 40], hsel);
    const v16h b0 = *(const v16h*)(Wb + (size_t)(wcol + m)      * C_ + k0 + 16 * hsel);
    const v16h b1 = *(const v16h*)(Wb + (size_t)(wcol + 16 + m) * C_ + k0 + 16 * hsel);
    acc0 = WMMA_F16(a, b0, acc0);
    acc1 = WMMA_F16(a, b1, acc1);

    if (more) *(v8h*)&Axs[cur ^ 1][xr * 40 + xc] = hnext;
    __syncthreads();                               // one barrier per chunk
    cur ^= 1;
  }

  // C/D layout: VGPR r, lanes 0-15 -> row r, lanes 16-31 -> row r+8; col = lane&15.
  if (which != 2) {                                // q/k: coalesced row-major f16
    _Float16* __restrict__ dst = (which == 0) ? qh : kh;
    for (int r = 0; r < 8; ++r) {
      const int rr = row0 + wrow + r + 8 * hsel;
      dst[(size_t)rr * H_ + wcol + m]      = (_Float16)acc0[r];
      dst[(size_t)rr * H_ + wcol + 16 + m] = (_Float16)acc1[r];
    }
  } else {                                         // v: transpose tile via LDS
    for (int r = 0; r < 8; ++r) {
      const int row = wrow + r + 8 * hsel;         // local token index 0..63
      Ots[(wcol + m)      * 72 + row] = (_Float16)acc0[r];
      Ots[(wcol + 16 + m) * 72 + row] = (_Float16)acc1[r];
    }
    __syncthreads();                               // uniform branch: safe
    const int bb = row0 >> 11, t0 = row0 & (T_ - 1);
    const int hh = tid >> 2, tc = (tid & 3) * 16;  // 64 h-rows x 4 chunks of 16
    *(v16h*)(vT + ((size_t)bb * H_ + hh) * T_ + t0 + tc) = *(const v16h*)&Ots[hh * 72 + tc];
  }
}

// ---------------------------------------------------------------------------
// Fused causal flash attention: one block per (batch, 64-query tile).
// 256 threads = 8 waves; wave w owns O rows (w&3)*16..+16, cols (w>>2)*32..+32.
// ---------------------------------------------------------------------------
__global__ __launch_bounds__(256) void attn_kernel(
    const _Float16* __restrict__ qh, const _Float16* __restrict__ kh,
    const _Float16* __restrict__ vT, float* __restrict__ out)
{
  __shared__ float    Sls[64 * 66];     // raw scores (padded stride 66)
  __shared__ _Float16 Pls[64 * 80];     // exp probabilities f16 (stride 80 = 160B)
  __shared__ float    alpha_ls[64];     // per-row rescale exp(m_old - m_new)
  __shared__ float    linv_ls[64];      // per-row 1/l at the final tile

  const int bb = blockIdx.x >> 5;                  // batch
  const int i0 = (blockIdx.x & 31) * 64;           // query tile base
  const int tid  = threadIdx.x;
  const int wave = tid >> 5, lane = tid & 31;
  const int m = lane & 15, hsel = lane >> 4;
  const int wrow = (wave & 3) * 16;
  const int wcol = (wave >> 2) * 32;

  // Q fragments for this wave's 16 rows, h-windows [0,32) and [32,64): pinned in VGPRs.
  const _Float16* qrow = qh + ((size_t)bb * T_ + i0 + wrow + m) * H_;
  const v16h qa0 = make_afrag(qrow,      hsel);
  const v16h qa1 = make_afrag(qrow + 32, hsel);

  v8f o0 = {}; v8f o1 = {};
  float mrun = -__builtin_inff();
  float lrun = 0.f;
  const float scale = 0.125f;                      // H^-0.5

  const int row4 = tid >> 2, q4 = (tid & 3) * 16;  // softmax: 4 threads per row

  for (int j0 = 0; j0 <= i0; j0 += 64) {
    // ---- S = Q K^T (16x32 per wave: two 16x16 tiles, K=64 as two k-steps) ----
    const _Float16* kb0 = kh + ((size_t)bb * T_ + j0 + wcol + m) * H_;
    const _Float16* kb1 = kb0 + 16 * H_;
    if (j0 + 64 <= i0) {                           // prefetch next k/v tiles
      __builtin_prefetch(kh + ((size_t)bb * T_ + j0 + 64 + wcol + m) * H_, 0, 1);
      __builtin_prefetch(vT + ((size_t)bb * H_ + wcol + m) * T_ + j0 + 64, 0, 1);
    }
    v8f s0 = {}; v8f s1 = {};
    s0 = WMMA_F16(qa0, *(const v16h*)(kb0 + 16 * hsel),      s0);
    s0 = WMMA_F16(qa1, *(const v16h*)(kb0 + 32 + 16 * hsel), s0);
    s1 = WMMA_F16(qa0, *(const v16h*)(kb1 + 16 * hsel),      s1);
    s1 = WMMA_F16(qa1, *(const v16h*)(kb1 + 32 + 16 * hsel), s1);

    for (int r = 0; r < 8; ++r) {
      Sls[(wrow + r + 8 * hsel) * 66 + wcol + m]      = s0[r];
      Sls[(wrow + r + 8 * hsel) * 66 + wcol + 16 + m] = s1[r];
    }
    __syncthreads();

    // ---- online softmax: 4 threads/row, butterfly reduce within quad ----
    {
      const int jlim = (j0 == i0) ? row4 : 63;     // causal: j0+col <= i0+row
      float rmax = -__builtin_inff();
      for (int j = q4; j < q4 + 16; ++j)
        if (j <= jlim) rmax = fmaxf(rmax, Sls[row4 * 66 + j]);
      rmax = fmaxf(rmax, __shfl_xor(rmax, 1));
      rmax = fmaxf(rmax, __shfl_xor(rmax, 2));
      rmax *= scale;
      const float mnew = fmaxf(mrun, rmax);
      const float a = __expf(mrun - mnew);
      float lsum = 0.f;
      for (int j = q4; j < q4 + 16; ++j) {
        float p = (j <= jlim) ? __expf(Sls[row4 * 66 + j] * scale - mnew) : 0.f;
        lsum += p;
        Pls[row4 * 80 + j] = (_Float16)p;
      }
      lsum += __shfl_xor(lsum, 1);
      lsum += __shfl_xor(lsum, 2);
      lrun = lrun * a + lsum;
      mrun = mnew;
      if ((tid & 3) == 0) {
        alpha_ls[row4] = a;
        if (j0 + 64 > i0) linv_ls[row4] = 1.f / lrun;  // last key tile
      }
    }
    __syncthreads();

    // ---- O = O*alpha + P V ----
    for (int r = 0; r < 8; ++r) {
      const float a = alpha_ls[wrow + r + 8 * hsel];
      o0[r] *= a; o1[r] *= a;
    }
    const _Float16* prow = &Pls[(wrow + m) * 80];
    const _Float16* vb0  = vT + ((size_t)bb * H_ + wcol + m) * T_ + j0;
    const _Float16* vb1  = vb0 + (size_t)16 * T_;
    for (int kc = 0; kc < 2; ++kc) {
      const v16h pa = make_afrag(prow + kc * 32, hsel);
      o0 = WMMA_F16(pa, *(const v16h*)(vb0 + kc * 32 + 16 * hsel), o0);
      o1 = WMMA_F16(pa, *(const v16h*)(vb1 + kc * 32 + 16 * hsel), o1);
    }
    // No barrier needed here: next iteration's S-write is fenced from this
    // iteration's S-reads by the post-softmax barrier; P/alpha are rewritten
    // only after the next pre-softmax barrier.
  }

  // ---- normalize and store f32 output [B,T,H] ----
  for (int r = 0; r < 8; ++r) {
    const int row = wrow + r + 8 * hsel;
    const float li = linv_ls[row];
    const size_t o = ((size_t)bb * T_ + i0 + row) * H_;
    out[o + wcol + m]      = o0[r] * li;
    out[o + wcol + 16 + m] = o1[r] * li;
  }
}

// ---------------------------------------------------------------------------
extern "C" void kernel_launch(void* const* d_in, const int* in_sizes, int n_in,
                              void* d_out, int out_size, void* d_ws, size_t ws_size,
                              hipStream_t stream) {
  const float* x  = (const float*)d_in[0];
  const float* Wq = (const float*)d_in[1];
  const float* Wk = (const float*)d_in[2];
  const float* Wv = (const float*)d_in[3];

  const size_t n_qkv = (size_t)B_ * T_ * H_;            // 1,048,576 elements each
  const size_t n_wt  = (size_t)3 * C_ * H_;             // 147,456 elements
  if (ws_size < (3 * n_qkv + n_wt) * sizeof(_Float16)) return;  // ~6.3 MB scratch

  _Float16* qh = (_Float16*)d_ws;
  _Float16* kh = qh + n_qkv;
  _Float16* vT = kh + n_qkv;
  _Float16* Wt = vT + n_qkv;

  // 0) one-time f32 -> f16 transposed weights (L2-resident for the GEMM)
  wcvt_kernel<<<dim3((3 * C_ * H_) / 256), 256, 0, stream>>>(Wq, Wk, Wv, Wt);

  // 1) q/k/v projections (v transposed), f32 -> f16
  dim3 pgrid((B_ * T_) / 64, 3);
  proj_kernel<<<pgrid, 256, 0, stream>>>(x, Wt, qh, kh, vT);

  // 2) fused causal flash attention
  dim3 agrid(B_ * (T_ / 64));
  attn_kernel<<<agrid, 256, 0, stream>>>(qh, kh, vT, (float*)d_out);
}
